// SwinTransformerBlock_76751065579901
// MI455X (gfx1250) — matrix operand order repaired and checked by reading, not compiled, not run on hardware
//
#include <hip/hip_runtime.h>
#include <hip/hip_bf16.h>

typedef __attribute__((ext_vector_type(16))) _Float16 v16h;
typedef __attribute__((ext_vector_type(8)))  _Float16 v8h;
typedef __attribute__((ext_vector_type(8)))  float    v8f;

#define WMMA_F16(a, b, c) \
  __builtin_amdgcn_wmma_f32_16x16x32_f16(false, (a), false, (b), (short)0, (c), false, false)

// ---------------------------------------------------------------------------
// Fragment loaders (CDNA5 ISA 7.12.2 layouts, wave32). Work for global or LDS.
// A: 16(M)x32(K) f16 row-major, ld in elements.
//   lane L: row=L&15, halves hold K in [kh*8,kh*8+8) and [16+kh*8,16+kh*8+8)
// B: from Bt (N x K row-major == column-major B). lane L: col=L&15,
//   halves hold K in [kh*8,kh*8+8) and [16+kh*8,16+kh*8+8) (same as A).
// ---------------------------------------------------------------------------
__device__ __forceinline__ v16h load_frag_a(const _Float16* base, int ld) {
  const int lane = threadIdx.x & 31;
  const int r    = lane & 15;
  const int kh   = (lane >> 4) << 3;
  const _Float16* p = base + (size_t)r * ld + kh;
  v8h lo = *(const v8h*)(p);
  v8h hi = *(const v8h*)(p + 16);
  v16h f;
#pragma unroll
  for (int i = 0; i < 8; ++i) { f[i] = lo[i]; f[i + 8] = hi[i]; }
  return f;
}

__device__ __forceinline__ v16h load_frag_b(const _Float16* bt, int ld) {
  const int lane = threadIdx.x & 31;
  const int c    = lane & 15;
  const int kb   = (lane >> 4) << 3;
  const _Float16* p = bt + (size_t)c * ld + kb;
  v8h lo = *(const v8h*)(p);
  v8h hi = *(const v8h*)(p + 16);
  v16h f;
#pragma unroll
  for (int i = 0; i < 8; ++i) { f[i] = lo[i]; f[i + 8] = hi[i]; }
  return f;
}

__device__ __forceinline__ float wave_sum(float v) {
#pragma unroll
  for (int m = 16; m >= 1; m >>= 1) v += __shfl_xor(v, m, 32);
  return v;
}

// ---------------------------------------------------------------------------
#define CB      64
#define CH      56
#define CC      128
#define CNW     64
#define CN      49
#define NPAD    64
#define CHD     32
#define CHID    512
#define MT      (CB * CH * CH)      // 200704 token rows
#define QK_SCALE 0.1767766952966369f

// Flat LDS half-buffer offsets (in halves) for k_winblock
#define OFF_XWIN 0          // [64][128]
#define OFF_QS   8192       // [64][128]  q*scale
#define OFF_KTS  16384      // [64][128]  k
#define OFF_VTS  24576      // [128][64]  v^T
#define OFF_OB   32768      // [64][128]  attention out
#define OFF_PB   40960      // [64][64]   probs f16
#define SMEMH_SZ 45056

// ---------------------------------------------------------------------------
// Weight transpose + f16 convert:  wt[n*K + k] = w[k*N + n]
// ---------------------------------------------------------------------------
__global__ void k_wT(const float* __restrict__ w, _Float16* __restrict__ wt, int K, int N) {
  int idx = blockIdx.x * 256 + threadIdx.x;
  if (idx >= K * N) return;
  int nn = idx / K, kk = idx - nn * K;
  wt[idx] = (_Float16)w[(size_t)kk * N + nn];
}

// ---------------------------------------------------------------------------
// Fused window block: LN1 + QKV + 4-head attention + proj + residual.
// One block per (image, window): 4096 blocks x 8 waves. ~105 KB LDS.
// ---------------------------------------------------------------------------
__global__ void __launch_bounds__(256)
k_winblock(const float* __restrict__ x, const float* __restrict__ mask,
           const float* __restrict__ ln1_g, const float* __restrict__ ln1_b,
           const _Float16* __restrict__ wqkvT, const _Float16* __restrict__ wprojT,
           const float* __restrict__ b_proj, float* __restrict__ x2) {
  __shared__ __align__(16) _Float16 smemh[SMEMH_SZ];   // 88 KB, see OFF_*
  __shared__ float Sb[NPAD][NPAD + 4];                 // 17 KB scores (fp32)

  // wave id as an SGPR: all derived per-wave control flow becomes scalar
  const int wave = __builtin_amdgcn_readfirstlane(threadIdx.x >> 5);
  const int lane = threadIdx.x & 31;
  const int col = lane & 15, kh = lane >> 4;
  const int wh = blockIdx.x;
  const int b = wh >> 6, w = wh & 63;

  __builtin_prefetch(wqkvT + (size_t)wave * 6144, 0, 1);   // global_prefetch_b8

  // ---- stage 1: roll(-3,-3) + window partition + LN1 -> xwin (f16) --------
#pragma unroll
  for (int rr = 0; rr < 8; ++rr) {
    const int n = wave * 8 + rr;                       // scalar
    _Float16* dst = smemh + OFF_XWIN + n * CC + lane * 4;
    if (n >= CN) {
      dst[0] = dst[1] = dst[2] = dst[3] = (_Float16)0.f;
      continue;
    }
    const int hs = (w >> 3) * 7 + n / 7;
    const int ws = (w & 7) * 7 + n % 7;
    const int hh = (hs + 3) % CH;                      // shifted[hs]=x[(hs+3)%56]
    const int ww = (ws + 3) % CH;
    const float* src = x + ((size_t)b * (CH * CH) + hh * CH + ww) * CC;
    float4 v = *(const float4*)(src + lane * 4);
    float s = wave_sum(v.x + v.y + v.z + v.w);
    const float mean = s * (1.f / CC);
    float dx = v.x - mean, dy = v.y - mean, dz = v.z - mean, dw = v.w - mean;
    float q = wave_sum(dx * dx + dy * dy + dz * dz + dw * dw);
    const float rstd = rsqrtf(q * (1.f / CC) + 1e-5f);
    const float4 gv = *(const float4*)(ln1_g + lane * 4);
    const float4 bv = *(const float4*)(ln1_b + lane * 4);
    dst[0] = (_Float16)(dx * rstd * gv.x + bv.x);
    dst[1] = (_Float16)(dy * rstd * gv.y + bv.y);
    dst[2] = (_Float16)(dz * rstd * gv.z + bv.z);
    dst[3] = (_Float16)(dw * rstd * gv.w + bv.w);
  }
  __syncthreads();

  // ---- stage 2: QKV GEMM (64x384, K=128) -> qs / kts / vts (branch-free) --
#pragma unroll
  for (int ci = 0; ci < 3; ++ci) {
    const int j0 = (wave * 3 + ci) * 16;               // scalar, 0..383
    const int s = j0 >> 7;                             // 0=q 1=k 2=v (scalar)
    const int hd = (j0 & 127) + col;                   // column within 128
    const float scl = (s == 0) ? QK_SCALE : 1.f;
    const int qkbase = OFF_QS + s * 8192;              // qs or kts
#pragma unroll
    for (int ri = 0; ri < 4; ++ri) {
      v8f c = {};
#pragma unroll
      for (int ks = 0; ks < 4; ++ks) {
        v16h a = load_frag_a(smemh + OFF_XWIN + ri * 16 * CC + ks * 32, CC);
        v16h bm = load_frag_b(wqkvT + (size_t)j0 * CC + ks * 32, CC);
        c = WMMA_F16(a, bm, c);
      }
#pragma unroll
      for (int i = 0; i < 8; ++i) {
        const int n = ri * 16 + i + 8 * kh;
        const int addr = (s == 2) ? (OFF_VTS + hd * NPAD + n)   // V^T scatter
                                  : (qkbase + n * CC + hd);     // q / k
        smemh[addr] = (_Float16)(c[i] * scl);
      }
    }
  }
  __syncthreads();

  // ---- stage 3: attention, heads sequential (Sb/Pb reused) ----------------
  for (int h = 0; h < 4; ++h) {
    // S = q k^T : 16 tiles, 2 per wave (K=32 -> single WMMA each)
#pragma unroll
    for (int u = 0; u < 2; ++u) {
      const int id = wave * 2 + u;
      const int ri = id >> 2, t = id & 3;
      v16h a = load_frag_a(smemh + OFF_QS + ri * 16 * CC + h * CHD, CC);
      v16h bm = load_frag_b(smemh + OFF_KTS + t * 16 * CC + h * CHD, CC);
      v8f c = {};
      c = WMMA_F16(a, bm, c);
#pragma unroll
      for (int i = 0; i < 8; ++i) Sb[ri * 16 + i + 8 * kh][t * 16 + col] = c[i];
    }
    __syncthreads();

    // masked softmax (fp32), one thread per row
    const int n = threadIdx.x;
    if (n < NPAD) {
      _Float16* prow = smemh + OFF_PB + n * NPAD;
      if (n < CN) {
        const float* mrow = mask + ((size_t)w * CN + n) * CN;
        float mx = -1e30f;
        for (int m = 0; m < CN; ++m) {
          float v = Sb[n][m] + mrow[m];
          Sb[n][m] = v;
          mx = fmaxf(mx, v);
        }
        float sum = 0.f;
        for (int m = 0; m < CN; ++m) {
          float e = __expf(Sb[n][m] - mx);
          Sb[n][m] = e;
          sum += e;
        }
        const float inv = 1.f / sum;
        for (int m = 0; m < CN; ++m) prow[m] = (_Float16)(Sb[n][m] * inv);
        for (int m = CN; m < NPAD; ++m) prow[m] = (_Float16)0.f;
      } else {
        for (int m = 0; m < NPAD; ++m) prow[m] = (_Float16)0.f;
      }
    }
    __syncthreads();

    // O = P @ V : 8 tiles, 1 per wave (K=64 -> 2 WMMA)
    {
      const int ri = wave >> 1, t = wave & 1;
      v8f c = {};
#pragma unroll
      for (int ks = 0; ks < 2; ++ks) {
        v16h a = load_frag_a(smemh + OFF_PB + ri * 16 * NPAD + ks * 32, NPAD);
        v16h bm = load_frag_b(smemh + OFF_VTS + (h * CHD + t * 16) * NPAD + ks * 32, NPAD);
        c = WMMA_F16(a, bm, c);
      }
#pragma unroll
      for (int i = 0; i < 8; ++i)
        smemh[OFF_OB + (ri * 16 + i + 8 * kh) * CC + h * CHD + t * 16 + col] =
            (_Float16)c[i];
    }
    __syncthreads();
  }

  // ---- stage 4: proj GEMM + bias + reverse-shift + residual -> x2 ---------
  {
    const int ri = wave >> 1;
#pragma unroll
    for (int cj = 0; cj < 4; ++cj) {
      const int j0 = ((wave & 1) * 4 + cj) * 16;       // scalar
      v8f c = {};
#pragma unroll
      for (int ks = 0; ks < 4; ++ks) {
        v16h a = load_frag_a(smemh + OFF_OB + ri * 16 * CC + ks * 32, CC);
        v16h bm = load_frag_b(wprojT + (size_t)j0 * CC + ks * 32, CC);
        c = WMMA_F16(a, bm, c);
      }
      const int jc = j0 + col;
      const float bv = b_proj[jc];
#pragma unroll
      for (int i = 0; i < 8; ++i) {
        const int n = ri * 16 + i + 8 * kh;
        if (n < CN) {
          const int hs = (w >> 3) * 7 + n / 7;
          const int ws = (w & 7) * 7 + n % 7;
          const int hh = (hs + 3) % CH;      // same pixel the block loaded (L2 hot)
          const int ww = (ws + 3) % CH;
          const size_t idx = ((size_t)b * (CH * CH) + hh * CH + ww) * CC + jc;
          x2[idx] = x[idx] + c[i] + bv;
        }
      }
    }
  }
}

// ---------------------------------------------------------------------------
// Fused LN2 + MLP(fc1 + exact GELU -> LDS -> fc2) + residual.
// 16-row stripe per block, 8 waves. ~28 KB LDS. grid = MT/16.
// ---------------------------------------------------------------------------
__global__ void __launch_bounds__(256)
k_mlp2(const float* __restrict__ x2, const float* __restrict__ ln2_g,
       const float* __restrict__ ln2_b, const _Float16* __restrict__ W1t,
       const float* __restrict__ b1, const _Float16* __restrict__ W2t,
       const float* __restrict__ b2, float* __restrict__ out) {
  __shared__ float xr[16][CC];                       // residual / LN input (8 KB)
  __shared__ __align__(16) _Float16 a16[16][CC];     // LN2 output f16 (4 KB)
  __shared__ __align__(16) _Float16 Hs[16][CHID];    // GELU(fc1) f16 (16 KB)
  const int wave = __builtin_amdgcn_readfirstlane(threadIdx.x >> 5);
  const int lane = threadIdx.x & 31;
  const int col = lane & 15, kh = lane >> 4;
  const size_t m0 = (size_t)blockIdx.x * 16;

  __builtin_prefetch(W1t + (size_t)wave * 8192, 0, 1);

  // ---- LN2 on 16 rows (2 rows per wave) -----------------------------------
#pragma unroll
  for (int rr = 0; rr < 2; ++rr) {
    const int r = wave * 2 + rr;                       // scalar
    const float* src = x2 + (m0 + r) * CC;
    float4 v = *(const float4*)(src + lane * 4);
    *(float4*)(&xr[r][lane * 4]) = v;
    float s = wave_sum(v.x + v.y + v.z + v.w);
    const float mean = s * (1.f / CC);
    float dx = v.x - mean, dy = v.y - mean, dz = v.z - mean, dw = v.w - mean;
    float q = wave_sum(dx * dx + dy * dy + dz * dz + dw * dw);
    const float rstd = rsqrtf(q * (1.f / CC) + 1e-5f);
    const float4 gv = *(const float4*)(ln2_g + lane * 4);
    const float4 bv = *(const float4*)(ln2_b + lane * 4);
    a16[r][lane * 4 + 0] = (_Float16)(dx * rstd * gv.x + bv.x);
    a16[r][lane * 4 + 1] = (_Float16)(dy * rstd * gv.y + bv.y);
    a16[r][lane * 4 + 2] = (_Float16)(dz * rstd * gv.z + bv.z);
    a16[r][lane * 4 + 3] = (_Float16)(dw * rstd * gv.w + bv.w);
  }
  __syncthreads();

  // ---- fc1 + bias + exact GELU -> Hs --------------------------------------
  v16h afr[4];
#pragma unroll
  for (int ks = 0; ks < 4; ++ks) afr[ks] = load_frag_a(&a16[0][ks * 32], CC);

#pragma unroll
  for (int t = 0; t < 4; ++t) {
    const int j0 = wave * 64 + t * 16;                 // scalar
    v8f c = {};
#pragma unroll
    for (int ks = 0; ks < 4; ++ks) {
      v16h bm = load_frag_b(W1t + (size_t)j0 * CC + ks * 32, CC);
      c = WMMA_F16(afr[ks], bm, c);
    }
    const int jc = j0 + col;
    const float bv = b1[jc];
#pragma unroll
    for (int i = 0; i < 8; ++i) {
      float v = c[i] + bv;
      v = 0.5f * v * (1.f + erff(v * 0.70710678118654752f));   // exact GELU
      Hs[i + 8 * kh][jc] = (_Float16)v;
    }
  }
  __syncthreads();

  // ---- fc2 + bias + residual -> out ---------------------------------------
  const int j0 = wave * 16;                            // scalar
  v8f c = {};
#pragma unroll
  for (int ks = 0; ks < 16; ++ks) {
    v16h a = load_frag_a(&Hs[0][ks * 32], CHID);
    v16h bm = load_frag_b(W2t + (size_t)j0 * CHID + ks * 32, CHID);
    c = WMMA_F16(a, bm, c);
  }
  const int jc = j0 + col;
  const float bv = b2[jc];
#pragma unroll
  for (int i = 0; i < 8; ++i) {
    const int r = i + 8 * kh;
    out[(m0 + r) * CC + jc] = xr[r][jc] + c[i] + bv;
  }
}

// ---------------------------------------------------------------------------
// Host launcher. Workspace: x2 fp32 (98 MiB) + f16 transposed weights.
// ---------------------------------------------------------------------------
extern "C" void kernel_launch(void* const* d_in, const int* in_sizes, int n_in,
                              void* d_out, int out_size, void* d_ws, size_t ws_size,
                              hipStream_t stream) {
  (void)in_sizes; (void)n_in; (void)out_size; (void)ws_size;
  const float* x      = (const float*)d_in[0];
  const float* mask   = (const float*)d_in[1];
  const float* ln1_g  = (const float*)d_in[2];
  const float* ln1_b  = (const float*)d_in[3];
  const float* w_qkv  = (const float*)d_in[4];
  const float* w_proj = (const float*)d_in[5];
  const float* b_proj = (const float*)d_in[6];
  const float* ln2_g  = (const float*)d_in[7];
  const float* ln2_b  = (const float*)d_in[8];
  const float* w1     = (const float*)d_in[9];
  const float* b1     = (const float*)d_in[10];
  const float* w2     = (const float*)d_in[11];
  const float* b2     = (const float*)d_in[12];
  float* out = (float*)d_out;

  char* ws = (char*)d_ws;
  const size_t MiB = 1ull << 20;
  float*    x2     = (float*)   (ws);                 // 98 MiB
  _Float16* wqkvT  = (_Float16*)(ws + 112 * MiB);     // 96 KiB
  _Float16* wprojT = (_Float16*)(ws + 113 * MiB);     // 32 KiB
  _Float16* w1T    = (_Float16*)(ws + 114 * MiB);     // 128 KiB
  _Float16* w2T    = (_Float16*)(ws + 115 * MiB);     // 128 KiB

  k_wT<<<(128 * 384 + 255) / 256, 256, 0, stream>>>(w_qkv,  wqkvT, 128, 384);
  k_wT<<<(128 * 128 + 255) / 256, 256, 0, stream>>>(w_proj, wprojT, 128, 128);
  k_wT<<<(128 * 512 + 255) / 256, 256, 0, stream>>>(w1,     w1T,   128, 512);
  k_wT<<<(512 * 128 + 255) / 256, 256, 0, stream>>>(w2,     w2T,   512, 128);

  // LN1 + QKV + attention + proj + residual, one block per window
  k_winblock<<<CB * CNW, 256, 0, stream>>>(x, mask, ln1_g, ln1_b,
                                           wqkvT, wprojT, b_proj, x2);

  // LN2 + MLP + residual
  k_mlp2<<<MT / 16, 256, 0, stream>>>(x2, ln2_g, ln2_b, w1T, b1, w2T, b2, out);
}
